// ParallelHSRULayer_80985903333783
// MI455X (gfx1250) — matrix operand order repaired and checked by reading, not compiled, not run on hardware
//
#include <hip/hip_runtime.h>
#include <cstdint>
#include <cstddef>

// ---------------------------------------------------------------------------
// Types for CDNA5 WMMA (wave32, v_wmma_f32_16x16x32_bf16)
// ---------------------------------------------------------------------------
typedef __attribute__((ext_vector_type(16))) __bf16 v16bf;
typedef __attribute__((ext_vector_type(8)))  float  v8f;

union Frag32 { uint4 u[2]; v16bf bf; };   // 32 bytes = 16 bf16
union Pack8  { __bf16 h[8]; uint4 u; };   // 16 bytes = 8 bf16

// Problem dims (fixed by the reference)
#define BB 16
#define SS 2048
#define INF 1024
#define HH 1024
#define MM (BB * SS)          // 32768 rows

// GEMM tiling
#define TM 128
#define TN 256
#define TK 32
#define LDP 40                // LDS row pitch in bf16 elements (80B, 16B-aligned, bank-spread)

// ---------------------------------------------------------------------------
// f32 -> bf16 elementwise convert (n divisible by 8)
// ---------------------------------------------------------------------------
__global__ __launch_bounds__(256)
void k_cvt_f32_bf16(const float* __restrict__ in, __bf16* __restrict__ out, long n)
{
    long i = ((long)blockIdx.x * blockDim.x + threadIdx.x) * 8;
    if (i >= n) return;
    float4 a = *(const float4*)(in + i);
    float4 b = *(const float4*)(in + i + 4);
    Pack8 p;
    p.h[0] = (__bf16)a.x; p.h[1] = (__bf16)a.y; p.h[2] = (__bf16)a.z; p.h[3] = (__bf16)a.w;
    p.h[4] = (__bf16)b.x; p.h[5] = (__bf16)b.y; p.h[6] = (__bf16)b.z; p.h[7] = (__bf16)b.w;
    *(uint4*)(out + i) = p.u;
}

// ---------------------------------------------------------------------------
// Transpose + convert: W (K x N, f32, row-major) -> WT (N x K, bf16, row-major)
// block = (32, 8), grid = (N/32, K/32)
// ---------------------------------------------------------------------------
__global__ __launch_bounds__(256)
void k_transpose_f32_bf16(const float* __restrict__ W, __bf16* __restrict__ WT,
                          int K, int N)
{
    __shared__ float tile[32][33];
    int n0 = blockIdx.x * 32;
    int k0 = blockIdx.y * 32;
    int tx = threadIdx.x;      // 0..31
    int ty = threadIdx.y;      // 0..7
#pragma unroll
    for (int r = 0; r < 32; r += 8)
        tile[ty + r][tx] = W[(size_t)(k0 + ty + r) * N + (n0 + tx)];
    __syncthreads();
#pragma unroll
    for (int r = 0; r < 32; r += 8)
        WT[(size_t)(n0 + ty + r) * K + (k0 + tx)] = (__bf16)tile[tx][ty + r];
}

// ---------------------------------------------------------------------------
// bf16 WMMA GEMM:  C[M,N] = A[M,K] * BT[N,K]^T + bias[N]   (optionally tanh)
// 256 threads = 8 waves; waves arranged 2 (M) x 4 (N); each wave: 64x64 tile
// = 4x4 fragments of 16x16, K-step 32 -> 16 v_wmma per step.
// LDS double-buffered and filled with GLOBAL_LOAD_ASYNC_TO_LDS_B128 issued one
// K-step ahead; completion via s_wait_asynccnt (ASYNCcnt, in-order loads).
// Steady-state loop body is branch-free (last iteration peeled).
// ---------------------------------------------------------------------------
__global__ __launch_bounds__(256)
void k_gemm_bf16(const __bf16* __restrict__ A, const __bf16* __restrict__ BT,
                 const float* __restrict__ bias, float* __restrict__ C,
                 int M, int N, int K, int do_tanh)
{
    __shared__ __align__(16) __bf16 As[2][TM * LDP];  // [m][k] pitch LDP
    __shared__ __align__(16) __bf16 Bs[2][TN * LDP];  // [n][k] pitch LDP

    const int tid   = threadIdx.x;
    const int wave  = tid >> 5;
    const int lane  = tid & 31;
    const int wm    = wave & 1;        // 0..1
    const int wn    = wave >> 1;       // 0..3
    const int lrow  = lane & 15;       // 0..15
    const int khalf = lane >> 4;       // 0..1

    const int m0 = blockIdx.x * TM;
    const int n0 = blockIdx.y * TN;

    // per-thread staging geometry (6 x 16B async chunks per thread per K-step)
    const int mA0 = tid >> 2,          cA0 = tid & 3;
    const int mA1 = (tid + 256) >> 2,  cA1 = (tid + 256) & 3;

    v8f acc[4][4] = {};

    const int nk = K / TK;

    // issue one K-tile's async global->LDS transfers into buffer `buf`
    auto issue_tile = [&](int kt, int buf) {
        const int k0 = kt * TK;
        {
            uint32_t l0 = (uint32_t)(uintptr_t)&As[buf][mA0 * LDP + cA0 * 8];
            uint64_t g0 = (uint64_t)(uintptr_t)(A + (size_t)(m0 + mA0) * K + k0 + cA0 * 8);
            asm volatile("global_load_async_to_lds_b128 %0, %1, off"
                         :: "v"(l0), "v"(g0) : "memory");
            uint32_t l1 = (uint32_t)(uintptr_t)&As[buf][mA1 * LDP + cA1 * 8];
            uint64_t g1 = (uint64_t)(uintptr_t)(A + (size_t)(m0 + mA1) * K + k0 + cA1 * 8);
            asm volatile("global_load_async_to_lds_b128 %0, %1, off"
                         :: "v"(l1), "v"(g1) : "memory");
        }
#pragma unroll
        for (int i = 0; i < 4; ++i) {
            int id = tid + i * 256;            // 0..1023
            int n  = id >> 2;
            int c  = id & 3;
            uint32_t l = (uint32_t)(uintptr_t)&Bs[buf][n * LDP + c * 8];
            uint64_t g = (uint64_t)(uintptr_t)(BT + (size_t)(n0 + n) * K + k0 + c * 8);
            asm volatile("global_load_async_to_lds_b128 %0, %1, off"
                         :: "v"(l), "v"(g) : "memory");
        }
    };

    // compute one K-step from buffer `buf`
    auto compute_tile = [&](int buf) {
        Frag32 a[4], b[4];
#pragma unroll
        for (int i = 0; i < 4; ++i) {
            int m = wm * 64 + i * 16 + lrow;
            // A lane: row m; K = khalf*8 + [0..7]  and  16 + khalf*8 + [0..7]
            a[i].u[0] = *(const uint4*)(&As[buf][m * LDP + khalf * 8]);
            a[i].u[1] = *(const uint4*)(&As[buf][m * LDP + 16 + khalf * 8]);
        }
#pragma unroll
        for (int j = 0; j < 4; ++j) {
            int n = wn * 64 + j * 16 + lrow;
            // B lane: col n; K = khalf*8 + [0..7]  and  16 + khalf*8 + [0..7]
            b[j].u[0] = *(const uint4*)(&Bs[buf][n * LDP + khalf * 8]);
            b[j].u[1] = *(const uint4*)(&Bs[buf][n * LDP + 16 + khalf * 8]);
        }
#pragma unroll
        for (int i = 0; i < 4; ++i)
#pragma unroll
            for (int j = 0; j < 4; ++j)
                acc[i][j] = __builtin_amdgcn_wmma_f32_16x16x32_bf16(
                    false, a[i].bf, false, b[j].bf,
                    (short)0, acc[i][j], false, false);
    };

    issue_tile(0, 0);

    // steady state: branch-free body
    for (int kt = 0; kt < nk - 1; ++kt) {
        const int buf = kt & 1;
        // prefetch next tile into the other buffer (read-safe: all waves
        // finished reading it at the trailing barrier of iteration kt-1)
        issue_tile(kt + 1, buf ^ 1);
        // in-order async completion: <=6 outstanding means tile kt landed
        asm volatile("s_wait_asynccnt 6" ::: "memory");
        __syncthreads();
        compute_tile(buf);
        __syncthreads();
    }

    // peeled last iteration
    asm volatile("s_wait_asynccnt 0" ::: "memory");
    __syncthreads();
    compute_tile((nk - 1) & 1);

    // ---- epilogue: C/D layout vgpr r -> M = r + 8*khalf, N = lrow ----
#pragma unroll
    for (int j = 0; j < 4; ++j) {
        int col = n0 + wn * 64 + j * 16 + lrow;
        float bv = bias[col];
#pragma unroll
        for (int i = 0; i < 4; ++i) {
            int rowBase = m0 + wm * 64 + i * 16 + khalf * 8;
#pragma unroll
            for (int r = 0; r < 8; ++r) {
                float v = acc[i][j][r] + bv;
                if (do_tanh) {
                    float t = fminf(fmaxf(v, -12.f), 12.f);
                    float e = __expf(2.f * t);
                    v = __fdividef(e - 1.f, e + 1.f);
                }
                C[(size_t)(rowBase + r) * N + col] = v;
            }
        }
    }
}

// ---------------------------------------------------------------------------
// Hysteresis leaky-integrate scan. One thread per (b,h) chain.
// comb[b,t,0:H] = v ; comb[b,t,H:2H] = s   (bf16)
// ---------------------------------------------------------------------------
__global__ __launch_bounds__(256)
void k_hsru_scan(const float* __restrict__ I, const float* __restrict__ leak_tau,
                 const float* __restrict__ thr, __bf16* __restrict__ comb)
{
    int idx = blockIdx.x * blockDim.x + threadIdx.x;   // 0..B*H-1
    int h = idx & (HH - 1);
    int b = idx >> 10;                                 // /HH
    float lt = leak_tau[h];
    float sp = (lt > 20.f) ? lt : log1pf(__expf(lt));  // softplus
    float alpha = __expf(-sp);
    float th = thr[h];

    const float* Ip = I + (size_t)b * SS * HH + h;
    __bf16* cp = comb + (size_t)b * SS * (2 * HH) + h;

    float v = 0.f, s = -1.f;
    for (int t = 0; t < SS; ++t) {
        v = fmaf(alpha, v, Ip[(size_t)t * HH]);
        s = (v > th) ? 1.f : ((v < -th) ? -1.f : s);
        cp[(size_t)t * (2 * HH)]      = (__bf16)v;
        cp[(size_t)t * (2 * HH) + HH] = (__bf16)s;
    }
}

// ---------------------------------------------------------------------------
// Launch
// inputs: x, W_in, b_in, leak_tau, flip_threshold, W_out, b_out
// ---------------------------------------------------------------------------
extern "C" void kernel_launch(void* const* d_in, const int* in_sizes, int n_in,
                              void* d_out, int out_size, void* d_ws, size_t ws_size,
                              hipStream_t stream)
{
    const float* x      = (const float*)d_in[0];
    const float* W_in   = (const float*)d_in[1];
    const float* b_in   = (const float*)d_in[2];
    const float* ltau   = (const float*)d_in[3];
    const float* thr    = (const float*)d_in[4];
    const float* W_out  = (const float*)d_in[5];
    const float* b_out  = (const float*)d_in[6];
    float* out = (float*)d_out;

    // workspace carve-up (all 256B aligned)
    char* ws = (char*)d_ws;
    __bf16* x_bf   = (__bf16*)(ws);                          // 16*2048*1024*2   = 64 MiB
    __bf16* WinT   = (__bf16*)(ws + (size_t)67108864);       // 1024*1024*2      =  2 MiB
    __bf16* WoutT  = (__bf16*)(ws + (size_t)69206016);       // 1024*2048*2      =  4 MiB
    float*  Icur   = (float*) (ws + (size_t)73400320);       // 32768*1024*4     = 128 MiB
    __bf16* comb   = (__bf16*)(ws + (size_t)207618048);      // 32768*2048*2     = 128 MiB
    // total = 341,835,776 bytes

    // 1) x -> bf16
    {
        long n = (long)MM * INF;                 // 33,554,432
        dim3 g((unsigned)(n / 8 / 256));
        k_cvt_f32_bf16<<<g, 256, 0, stream>>>(x, x_bf, n);
    }
    // 2) W_in^T, W_out^T -> bf16
    {
        dim3 blk(32, 8);
        dim3 g1(HH / 32, INF / 32);              // W_in: K=1024, N=1024
        k_transpose_f32_bf16<<<g1, blk, 0, stream>>>(W_in, WinT, INF, HH);
        dim3 g2(HH / 32, (2 * HH) / 32);         // W_out: K=2048, N=1024
        k_transpose_f32_bf16<<<g2, blk, 0, stream>>>(W_out, WoutT, 2 * HH, HH);
    }
    // 3) I = x @ W_in + b_in
    {
        dim3 g(MM / TM, HH / TN);                // 256 x 4
        k_gemm_bf16<<<g, 256, 0, stream>>>(x_bf, WinT, b_in, Icur,
                                           MM, HH, INF, /*tanh=*/0);
    }
    // 4) scan -> comb (bf16, [v | s])
    {
        dim3 g((BB * HH) / 256);                 // 64
        k_hsru_scan<<<g, 256, 0, stream>>>(Icur, ltau, thr, comb);
    }
    // 5) out = tanh(comb @ W_out + b_out)
    {
        dim3 g(MM / TM, HH / TN);
        k_gemm_bf16<<<g, 256, 0, stream>>>(comb, WoutT, b_out, out,
                                           MM, HH, 2 * HH, /*tanh=*/1);
    }
    (void)in_sizes; (void)n_in; (void)out_size; (void)ws_size;
}
